// CausalMultiHeadSelfAttention_35485019799578
// MI455X (gfx1250) — compile-verified, matrix operand order checked by reading, not run on hardware
//
#include <hip/hip_runtime.h>
#include <hip/hip_bf16.h>

// ---------------------------------------------------------------------------
// Causal MHSA forward for MI455X (gfx1250), bf16 WMMA path.
// B=4, T=2048, C=1024, H=16, D=64.
// ---------------------------------------------------------------------------

typedef __bf16 bf16_t;
typedef __attribute__((ext_vector_type(16))) __bf16 v16bf;
typedef __attribute__((ext_vector_type(8)))  float  v8f;

#define B_ 4
#define T_ 2048
#define C_ 1024
#define H_ 16
#define D_ 64

__device__ __forceinline__ v8f vzero8() {
  v8f z = {0.f, 0.f, 0.f, 0.f, 0.f, 0.f, 0.f, 0.f};
  return z;
}

// Load a 16x32 bf16 WMMA fragment from a row-major matrix with leading dim
// `ld` (elements). Produces the ISA-documented 16-bit A layout:
//   lanes 0-15 : row = lane,    halves K 0..7 then 16..23
//   lanes 16-31: row = lane-16, halves K 8..15 then 24..31
// The same loader yields a valid B (KxN) fragment when `base` points at a
// row-major N x K matrix (i.e. B = source^T).
__device__ __forceinline__ v16bf load_frag(const bf16_t* __restrict__ base, int ld) {
  const int lane = threadIdx.x & 31;
  const int row  = lane & 15;
  const int k0   = (lane >> 4) << 3;  // 0 or 8
  const bf16_t* p = base + (size_t)row * ld + k0;
  v16bf f;
#pragma unroll
  for (int i = 0; i < 8; ++i) {
    f[i]     = p[i];       // K = k0 + 0..7
    f[8 + i] = p[16 + i];  // K = k0 + 16..23
  }
  return f;
}

__device__ __forceinline__ v8f wmma_bf16(v16bf a, v16bf b, v8f c) {
  return __builtin_amdgcn_wmma_f32_16x16x32_bf16(
      /*neg_a=*/false, a, /*neg_b=*/false, b,
      /*c_mod=*/(short)0, c, /*reuse_a=*/false, /*reuse_b=*/false);
}

// Async-copy a contiguous 4 KB block from global memory into LDS using the
// CDNA5 async-to-LDS path (tracked with ASYNCcnt). One wave: 8 x b128 per
// lane-row, 32 lanes x 16 B = 512 B per instruction.
__device__ __forceinline__ void async_copy_4k(unsigned lds_off,
                                              const bf16_t* __restrict__ src) {
  const int lane = threadIdx.x & 31;
  const unsigned lo = lds_off + (unsigned)(lane * 16);
  uint64_t ga = (uint64_t)(size_t)src + (uint64_t)(lane * 16);
#pragma unroll
  for (int it = 0; it < 8; ++it) {
    asm volatile("global_load_async_to_lds_b128 %0, %1, off"
                 :
                 : "v"(lo + (unsigned)(it * 512)), "v"(ga + (uint64_t)(it * 512))
                 : "memory");
  }
}

// ---------------------------------------------------------------------------
// f32 -> bf16 cast
// ---------------------------------------------------------------------------
__global__ void cast_f32_bf16(const float* __restrict__ src,
                              bf16_t* __restrict__ dst, int n) {
  int i = blockIdx.x * blockDim.x + threadIdx.x;
  const int stride = gridDim.x * blockDim.x;
  for (; i < n; i += stride) dst[i] = (bf16_t)src[i];
}

// ---------------------------------------------------------------------------
// Projection: OUT(M=B*T, N=C) = A(M,C) @ W^T + bias.
// One wave computes a 16x64 tile; k-loop is double-buffered so fragment loads
// for k+32 overlap the WMMAs for k. MODE:
//   0 -> bf16 out, (B,H,T,D) layout        (Q and K)
//   2 -> bf16 out, (B,H,D,T) layout        (V, transposed for PV fragments)
//   3 -> f32  out, row-major (B*T, C)      (final output projection)
// ---------------------------------------------------------------------------
template <int MODE>
__global__ __launch_bounds__(32)
void proj_kernel(const bf16_t* __restrict__ A, const bf16_t* __restrict__ W,
                 const float* __restrict__ bias, bf16_t* __restrict__ outb,
                 float* __restrict__ outf) {
  const int n0   = blockIdx.x * 64;
  const int m0   = blockIdx.y * 16;
  const int lane = threadIdx.x & 31;

  const bf16_t* __restrict__ Arow = A + (size_t)m0 * C_;

  v8f acc[4];
#pragma unroll
  for (int t = 0; t < 4; ++t) acc[t] = vzero8();

  // prologue: fragments for k0 = 0
  v16bf a_cur = load_frag(Arow, C_);
  v16bf b_cur[4];
#pragma unroll
  for (int t = 0; t < 4; ++t)
    b_cur[t] = load_frag(W + (size_t)(n0 + 16 * t) * C_, C_);

  for (int k0 = 0; k0 < C_ - 32; k0 += 32) {
    const int kn = k0 + 32;
    __builtin_prefetch(Arow + kn + 32, 0, 3);
    __builtin_prefetch(W + (size_t)n0 * C_ + kn + 32, 0, 3);
    // load next-step fragments (independent of current WMMAs)
    v16bf a_nxt = load_frag(Arow + kn, C_);
    v16bf b_nxt[4];
#pragma unroll
    for (int t = 0; t < 4; ++t)
      b_nxt[t] = load_frag(W + (size_t)(n0 + 16 * t) * C_ + kn, C_);
    // consume current fragments
#pragma unroll
    for (int t = 0; t < 4; ++t) acc[t] = wmma_bf16(a_cur, b_cur[t], acc[t]);
    a_cur = a_nxt;
#pragma unroll
    for (int t = 0; t < 4; ++t) b_cur[t] = b_nxt[t];
  }
  // tail
#pragma unroll
  for (int t = 0; t < 4; ++t) acc[t] = wmma_bf16(a_cur, b_cur[t], acc[t]);

  // C/D layout: lane holds column n = n0+16t+(lane&15); VGPR v holds row
  // m = m0 + v + 8*(lane>>4).
  const int nl = lane & 15;
  const int mh = (lane >> 4) << 3;
#pragma unroll
  for (int t = 0; t < 4; ++t) {
    const int n = n0 + t * 16 + nl;
    const float bv = bias[n];
#pragma unroll
    for (int v = 0; v < 8; ++v) {
      const int m = m0 + mh + v;
      const float val = acc[t][v] + bv;
      if (MODE == 3) {
        outf[(size_t)m * C_ + n] = val;
      } else {
        const int bb = m >> 11;        // m / T
        const int tt = m & (T_ - 1);   // m % T
        const int h  = n >> 6;         // n / D
        const int d  = n & (D_ - 1);   // n % D
        if (MODE == 0) {
          outb[(((size_t)bb * H_ + h) * T_ + tt) * D_ + d] = (bf16_t)val;
        } else {  // MODE == 2: transposed V
          outb[(((size_t)bb * H_ + h) * D_ + d) * T_ + tt] = (bf16_t)val;
        }
      }
    }
  }
}

// ---------------------------------------------------------------------------
// Flash attention: one wave per 16-query tile per (b,h). 32 keys per step.
// K tiles (32x64 bf16 = 4 KB, contiguous) are double-buffered into LDS with
// global_load_async_to_lds_b128 / s_wait_asynccnt. Softmax uses all 32 lanes.
// Q,K: (B,H,T,D) bf16.  Vt: (B,H,D,T) bf16.  Ob: (B,T,C) bf16.
// ---------------------------------------------------------------------------
__global__ __launch_bounds__(32)
void attn_kernel(const bf16_t* __restrict__ Q, const bf16_t* __restrict__ K,
                 const bf16_t* __restrict__ Vt, bf16_t* __restrict__ Ob) {
  __shared__ bf16_t Ktile[2][32 * D_];   // 2 x 4 KB double buffer
  __shared__ float  S_lds[16][32];
  __shared__ bf16_t P_lds[16][32];
  __shared__ float  mrow[16];
  __shared__ float  lrow[16];
  __shared__ float  arow[16];
  __shared__ float  pmax[2][16];
  __shared__ float  psum[2][16];

  const int lane = threadIdx.x & 31;
  const int q0   = blockIdx.x * 16;
  const int bh   = blockIdx.y;

  const bf16_t* __restrict__ Qb = Q  + (size_t)bh * T_ * D_;
  const bf16_t* __restrict__ Kb = K  + (size_t)bh * T_ * D_;
  const bf16_t* __restrict__ Vb = Vt + (size_t)bh * D_ * T_;

  const float scale  = 0.125f;   // 1/sqrt(D)
  const float NEGBIG = -1.0e30f; // finite "-inf" (avoids inf-inf NaN)

  const unsigned kt_base[2] = {(unsigned)(size_t)&Ktile[0][0],
                               (unsigned)(size_t)&Ktile[1][0]};

  // Q fragments (reused across all key steps): rows q0..q0+15, D split 0/32.
  const v16bf qa0 = load_frag(Qb + (size_t)q0 * D_ + 0,  D_);
  const v16bf qa1 = load_frag(Qb + (size_t)q0 * D_ + 32, D_);

  v8f acc[4];
#pragma unroll
  for (int t = 0; t < 4; ++t) acc[t] = vzero8();

  if (lane < 16) { mrow[lane] = NEGBIG; lrow[lane] = 0.f; }
  __syncthreads();

  // prime the async pipeline with the first K tile
  async_copy_4k(kt_base[0], Kb);

  int buf = 0;
  for (int j0 = 0; j0 <= q0 + 15; j0 += 32) {
    // issue next tile's async copies before waiting on the current one
    const bool has_next = (j0 + 32) <= (q0 + 15);
    if (has_next) async_copy_4k(kt_base[buf ^ 1], Kb + (size_t)(j0 + 32) * D_);
    // async loads complete in order: leaving <=8 outstanding means the
    // current tile (issued first) has fully landed in LDS.
    if (has_next) {
      asm volatile("s_wait_asynccnt 0x8" ::: "memory");
    } else {
      asm volatile("s_wait_asynccnt 0x0" ::: "memory");
    }
    __syncthreads();

    const bf16_t* Kt = &Ktile[buf][0];

    // ---- S = Q @ K^T for 32 keys (two 16x16 tiles) -> LDS ----
#pragma unroll
    for (int jj = 0; jj < 32; jj += 16) {
      const v16bf kb0 = load_frag(Kt + (size_t)jj * D_ + 0,  D_);
      const v16bf kb1 = load_frag(Kt + (size_t)jj * D_ + 32, D_);
      v8f s = wmma_bf16(qa0, kb0, vzero8());
      s = wmma_bf16(qa1, kb1, s);
      const int n  = lane & 15;
      const int mh = (lane >> 4) << 3;
#pragma unroll
      for (int v = 0; v < 8; ++v) S_lds[mh + v][jj + n] = s[v];
    }
    __syncthreads();

    // ---- masked online softmax, all 32 lanes: (row, 16-column half) ----
    {
      const int r  = lane & 15;
      const int hi = lane >> 4;
      const int ch = hi << 4;  // column base 0 or 16
      float sv[16];
      float mpart = NEGBIG;
#pragma unroll
      for (int c = 0; c < 16; ++c) {
        const int col = ch + c;
        const float s = (j0 + col <= q0 + r) ? S_lds[r][col] * scale : NEGBIG;
        sv[c] = s;
        mpart = fmaxf(mpart, s);
      }
      pmax[hi][r] = mpart;
      __syncthreads();
      const float mold = mrow[r];
      const float mnew = fmaxf(mold, fmaxf(pmax[0][r], pmax[1][r]));
      float sum = 0.f;
#pragma unroll
      for (int c = 0; c < 16; ++c) {
        const float p = __expf(sv[c] - mnew);  // masked -> exp(~-1e30) = 0
        P_lds[r][ch + c] = (bf16_t)p;
        sum += p;
      }
      psum[hi][r] = sum;
      __syncthreads();
      if (lane < 16) {
        const float alpha = __expf(mold - mnew);
        lrow[r] = alpha * lrow[r] + psum[0][r] + psum[1][r];
        mrow[r] = mnew;
        arow[r] = alpha;
      }
      __syncthreads();
    }

    // ---- rescale accumulators by per-row alpha ----
    const int mh = (lane >> 4) << 3;
#pragma unroll
    for (int v = 0; v < 8; ++v) {
      const float a = arow[mh + v];
      acc[0][v] *= a; acc[1][v] *= a; acc[2][v] *= a; acc[3][v] *= a;
    }

    // ---- O += P(16x32) @ V(32x64); Vt is (D,T) so B frags use fast loader ----
    const v16bf pa = load_frag(&P_lds[0][0], 32);
#pragma unroll
    for (int t = 0; t < 4; ++t) {
      const v16bf vbf = load_frag(Vb + (size_t)(t * 16) * T_ + j0, T_);
      acc[t] = wmma_bf16(pa, vbf, acc[t]);
    }
    __syncthreads();  // protect S_lds/P_lds before next iteration

    buf ^= 1;
  }

  // ---- normalize by 1/l and store concat-head bf16 (B,T,C) ----
  const int nl = lane & 15;
  const int mh = (lane >> 4) << 3;
  const int bb = bh >> 4;   // bh / H
  const int h  = bh & 15;   // bh % H
#pragma unroll
  for (int v = 0; v < 8; ++v) {
    const int m = mh + v;
    const float linv = 1.0f / lrow[m];
    const size_t rowbase = ((size_t)bb * T_ + (q0 + m)) * C_ + (size_t)h * D_;
#pragma unroll
    for (int t = 0; t < 4; ++t) {
      Ob[rowbase + t * 16 + nl] = (bf16_t)(acc[t][v] * linv);
    }
  }
}

// ---------------------------------------------------------------------------
// Launch
// ---------------------------------------------------------------------------
extern "C" void kernel_launch(void* const* d_in, const int* in_sizes, int n_in,
                              void* d_out, int out_size, void* d_ws, size_t ws_size,
                              hipStream_t stream) {
  (void)in_sizes; (void)n_in; (void)out_size; (void)ws_size;

  const float* x  = (const float*)d_in[0];
  const float* Wq = (const float*)d_in[1];
  const float* bq = (const float*)d_in[2];
  const float* Wk = (const float*)d_in[3];
  const float* bk = (const float*)d_in[4];
  const float* Wv = (const float*)d_in[5];
  const float* bv = (const float*)d_in[6];
  const float* Wo = (const float*)d_in[7];
  const float* bo = (const float*)d_in[8];
  float* out = (float*)d_out;

  char* ws = (char*)d_ws;
  size_t off = 0;
  auto alloc = [&](size_t bytes) -> char* {
    char* p = ws + off;
    off += (bytes + 255) & ~(size_t)255;
    return p;
  };

  const size_t xN = (size_t)B_ * T_ * C_;
  const size_t wN = (size_t)C_ * C_;

  bf16_t* xb  = (bf16_t*)alloc(xN * sizeof(bf16_t));
  bf16_t* Wqb = (bf16_t*)alloc(wN * sizeof(bf16_t));
  bf16_t* Wkb = (bf16_t*)alloc(wN * sizeof(bf16_t));
  bf16_t* Wvb = (bf16_t*)alloc(wN * sizeof(bf16_t));
  bf16_t* Wob = (bf16_t*)alloc(wN * sizeof(bf16_t));
  bf16_t* Qb  = (bf16_t*)alloc(xN * sizeof(bf16_t));  // (B,H,T,D)
  bf16_t* Kb  = (bf16_t*)alloc(xN * sizeof(bf16_t));  // (B,H,T,D)
  bf16_t* Vtb = (bf16_t*)alloc(xN * sizeof(bf16_t));  // (B,H,D,T)
  bf16_t* Obf = (bf16_t*)alloc(xN * sizeof(bf16_t));  // (B,T,C)

  cast_f32_bf16<<<2048, 256, 0, stream>>>(x,  xb,  (int)xN);
  cast_f32_bf16<<<512,  256, 0, stream>>>(Wq, Wqb, (int)wN);
  cast_f32_bf16<<<512,  256, 0, stream>>>(Wk, Wkb, (int)wN);
  cast_f32_bf16<<<512,  256, 0, stream>>>(Wv, Wvb, (int)wN);
  cast_f32_bf16<<<512,  256, 0, stream>>>(Wo, Wob, (int)wN);

  dim3 pg(C_ / 64, (B_ * T_) / 16);  // (16, 512)
  proj_kernel<0><<<pg, 32, 0, stream>>>(xb, Wqb, bq, Qb,  nullptr);
  proj_kernel<0><<<pg, 32, 0, stream>>>(xb, Wkb, bk, Kb,  nullptr);
  proj_kernel<2><<<pg, 32, 0, stream>>>(xb, Wvb, bv, Vtb, nullptr);

  dim3 ag(T_ / 16, B_ * H_);  // (128, 64)
  attn_kernel<<<ag, 32, 0, stream>>>(Qb, Kb, Vtb, Obf);

  proj_kernel<3><<<pg, 32, 0, stream>>>(Obf, Wob, bo, nullptr, out);
}
